// DecoderBlock_90649579750162
// MI455X (gfx1250) — matrix operand order repaired and checked by reading, compiled, not run on hardware
//
#include <hip/hip_runtime.h>
#include <hip/hip_bf16.h>

// ---------------------------------------------------------------------------
// MI455X (gfx1250) decoder block.
// - All GEMMs + attention on V_WMMA_F32_16X16X32_BF16 (wave32, fp32 accum).
// - GEMM double-buffers LDS tiles with GLOBAL_LOAD_ASYNC_TO_LDS_B128 and
//   s_wait_asynccnt (CDNA5 async copy path).
// - Flash-style attention avoids the 512MB score tensor.
// ---------------------------------------------------------------------------

typedef __attribute__((ext_vector_type(16))) __bf16 bf16x16;
typedef __attribute__((ext_vector_type(8)))  __bf16 bf16x8;
typedef __attribute__((ext_vector_type(4)))  __bf16 bf16x4;
typedef __attribute__((ext_vector_type(8)))  float  floatx8;

#define H_DIM   1024
#define NHEAD   16
#define HEAD_D  64
#define DFF_DIM 4096
#define BATCH   2
#define SEQ     2048
#define SEQ_I   256
#define M_X     (BATCH * SEQ)    // 4096
#define M_IMG   (BATCH * SEQ_I)  // 512

// ---------------- WMMA fragment helpers (ISA 7.12.2 layouts) ----------------
// A (16x32 bf16): lane L holds row M=L%16; element e -> K = (e/8)*16 + (L/16)*8 + e%8
__device__ __forceinline__ bf16x16 frag_a(const __bf16* rowp, int half) {
    bf16x8 lo = *(const bf16x8*)(rowp + half * 8);
    bf16x8 hi = *(const bf16x8*)(rowp + 16 + half * 8);
    bf16x16 a;
#pragma unroll
    for (int e = 0; e < 8; ++e) { a[e] = lo[e]; a[e + 8] = hi[e]; }
    return a;
}
// B (32x16 bf16): lane L holds col N=L%16; element e -> K = (e/8)*16 + (L/16)*8 + e%8
__device__ __forceinline__ bf16x16 frag_b(const __bf16* rowp, int half) {
    bf16x8 lo = *(const bf16x8*)(rowp + half * 8);
    bf16x8 hi = *(const bf16x8*)(rowp + 16 + half * 8);
    bf16x16 b;
#pragma unroll
    for (int e = 0; e < 8; ++e) { b[e] = lo[e]; b[e + 8] = hi[e]; }
    return b;
}
__device__ __forceinline__ floatx8 wmma_bf16(bf16x16 a, bf16x16 b, floatx8 c) {
    return __builtin_amdgcn_wmma_f32_16x16x32_bf16(false, a, false, b,
                                                   (short)0, c, false, false);
}

// ------------------- CDNA5 async global->LDS copy (16B/lane) ----------------
// Generic shared pointer: low 32 bits are the LDS byte offset (flat aperture).
__device__ __forceinline__ void async_copy16(const __bf16* gptr, __bf16* lptr) {
    unsigned lds = (unsigned)(size_t)lptr;
    asm volatile("global_load_async_to_lds_b128 %0, %1, off"
                 :: "v"(lds), "v"(gptr) : "memory");
}
__device__ __forceinline__ void wait_async0() {
    asm volatile("s_wait_asynccnt 0x0" ::: "memory");
}

// ------------------------------- LayerNorm ----------------------------------
template <bool SCALE_IN>
__global__ __launch_bounds__(256)
void layernorm_kernel(const float* __restrict__ xin, float* __restrict__ resid_out,
                      __bf16* __restrict__ nx, const float* __restrict__ g,
                      const float* __restrict__ bta) {
    __shared__ float s1[256], s2[256];
    const int row = blockIdx.x;
    const float* xr = xin + (size_t)row * H_DIM;
    float v[4];
    float sum = 0.f, sq = 0.f;
#pragma unroll
    for (int i = 0; i < 4; ++i) {
        float t = xr[threadIdx.x + i * 256];
        if (SCALE_IN) t *= 0.1f;
        v[i] = t; sum += t; sq += t * t;
    }
    s1[threadIdx.x] = sum; s2[threadIdx.x] = sq;
    __syncthreads();
    for (int off = 128; off > 0; off >>= 1) {
        if (threadIdx.x < off) {
            s1[threadIdx.x] += s1[threadIdx.x + off];
            s2[threadIdx.x] += s2[threadIdx.x + off];
        }
        __syncthreads();
    }
    const float mean = s1[0] * (1.f / H_DIM);
    const float var  = s2[0] * (1.f / H_DIM) - mean * mean;
    const float rstd = rsqrtf(var + 1e-5f);
#pragma unroll
    for (int i = 0; i < 4; ++i) {
        const int c = threadIdx.x + i * 256;
        const float t = v[i];
        if (SCALE_IN) resid_out[(size_t)row * H_DIM + c] = t;
        const float y = (t - mean) * rstd * g[c] + bta[c];
        nx[(size_t)row * H_DIM + c] = (__bf16)y;
    }
}

__global__ __launch_bounds__(256)
void scale_bf16_kernel(const float* __restrict__ in, __bf16* __restrict__ out, int n) {
    const int i = blockIdx.x * 256 + threadIdx.x;
    if (i < n) out[i] = (__bf16)(in[i] * 0.1f);
}

// ----------------- Weight transform: W[K][N] f32 -> Wt[N][K] bf16 -----------
// One-time per launch; makes both GEMM operands straight bf16 async copies.
__global__ __launch_bounds__(256)
void transpose_cvt_kernel(const float* __restrict__ W, __bf16* __restrict__ Wt,
                          int K, int N) {
    __shared__ float t[32][33];
    const int k0 = blockIdx.y * 32, n0 = blockIdx.x * 32;
    const int tx = threadIdx.x & 31, ty = threadIdx.x >> 5;   // 8 rows of 32
#pragma unroll
    for (int i = 0; i < 4; ++i)
        t[ty + 8 * i][tx] = W[(size_t)(k0 + ty + 8 * i) * N + n0 + tx];
    __syncthreads();
#pragma unroll
    for (int i = 0; i < 4; ++i)
        Wt[(size_t)(n0 + ty + 8 * i) * K + k0 + tx] = (__bf16)t[tx][ty + 8 * i];
}

// ------------------------------ bf16 WMMA GEMM ------------------------------
// C[M,N] = epilogue(A_bf16[M,K] @ Wt_bf16[N][K]^T + bias).
// 256 thr = 8 waves; block tile 128x128; wave tile 32x64 (2x4 frags); K-step 32.
// LDS tiles double-buffered via async global->LDS; 96B row stride (16B-aligned).
#define BM  128
#define BN  128
#define BK  32
#define AST 48   // LDS row stride in elements (96B, multiple of 16B)

// Stage a 128x32 bf16 tile: 512 16B chunks over 256 threads (2 each).
__device__ __forceinline__ void stage_tile(const __bf16* __restrict__ G,
                                           __bf16 (*L)[AST], int tid, int ldg) {
#pragma unroll
    for (int i = 0; i < 2; ++i) {
        const int c  = tid + i * 256;    // 0..511
        const int r  = c >> 2;           // 4 chunks per row
        const int kc = (c & 3) << 3;     // element offset (8 elems per chunk)
        async_copy16(G + (size_t)r * ldg + kc, &L[r][kc]);
    }
}

template <bool GELU_, bool RESID, bool OUTBF, bool OUTF>
__global__ __launch_bounds__(256)
void gemm_bf16_kernel(const __bf16* __restrict__ A, const __bf16* __restrict__ Wt,
                      const float* __restrict__ bias, const float* __restrict__ resid,
                      __bf16* __restrict__ outB, float* __restrict__ outF,
                      int M, int N, int K, float fscale) {
    __shared__ __bf16 As[2][BM][AST];
    __shared__ __bf16 Bs[2][BN][AST];
    const int tid  = threadIdx.x;
    const int wave = tid >> 5, lane = tid & 31;
    const int half = lane >> 4, l16 = lane & 15;
    const int wr = wave >> 1, wc = wave & 1;        // wave tile: rows wr*32, cols wc*64
    const int rowBase = blockIdx.y * BM;
    const int colBase = blockIdx.x * BN;
    const __bf16* Ag = A  + (size_t)rowBase * K;
    const __bf16* Bg = Wt + (size_t)colBase * K;

    floatx8 acc[2][4] = {};

    stage_tile(Ag, As[0], tid, K);
    stage_tile(Bg, Bs[0], tid, K);
    wait_async0();
    __syncthreads();

    int cur = 0;
    for (int k0 = 0; k0 < K; k0 += BK) {
        const int nxt = cur ^ 1;
        if (k0 + BK < K) {                 // prefetch next tiles while computing
            stage_tile(Ag + k0 + BK, As[nxt], tid, K);
            stage_tile(Bg + k0 + BK, Bs[nxt], tid, K);
        }
        bf16x16 af[2], bw[4];
#pragma unroll
        for (int mi = 0; mi < 2; ++mi)
            af[mi] = frag_a(&As[cur][wr * 32 + mi * 16 + l16][0], half);
#pragma unroll
        for (int ni = 0; ni < 4; ++ni)
            bw[ni] = frag_b(&Bs[cur][wc * 64 + ni * 16 + l16][0], half);
#pragma unroll
        for (int mi = 0; mi < 2; ++mi)
#pragma unroll
            for (int ni = 0; ni < 4; ++ni)
                acc[mi][ni] = wmma_bf16(af[mi], bw[ni], acc[mi][ni]);
        wait_async0();
        __syncthreads();
        cur = nxt;
    }
    // Epilogue. C layout: element j -> M = 8*half + j, N = l16.
#pragma unroll
    for (int mi = 0; mi < 2; ++mi) {
#pragma unroll
        for (int ni = 0; ni < 4; ++ni) {
            const int col = colBase + wc * 64 + ni * 16 + l16;
            const float bb = bias[col];
#pragma unroll
            for (int j = 0; j < 8; ++j) {
                const int row = rowBase + wr * 32 + mi * 16 + half * 8 + j;
                float v = acc[mi][ni][j] + bb;
                if (GELU_) v = 0.5f * v * (1.f + erff(v * 0.70710678118654752f));
                if (RESID) v += resid[(size_t)row * N + col];
                v *= fscale;
                if (OUTF)  outF[(size_t)row * N + col] = v;
                if (OUTBF) outB[(size_t)row * N + col] = (__bf16)v;
            }
        }
    }
}

// --------------------------- Flash attention (WMMA) -------------------------
#define FQ 64
#define FK 64
#define FST 72   // padded stride (144B rows, 16B-aligned)
__global__ __launch_bounds__(256)
void flash_attn_kernel(const __bf16* __restrict__ Q, const __bf16* __restrict__ Kb,
                       const __bf16* __restrict__ Vb, __bf16* __restrict__ O,
                       int Lq, int Lk, float scale) {
    __shared__ __bf16 Qs[FQ][FST];
    __shared__ __bf16 Ks[FK][FST];
    __shared__ __bf16 Vt[HEAD_D][FST];   // [hd][key]
    __shared__ float  Ss[FQ][FST];
    __shared__ __bf16 Ps[FQ][FST];
    __shared__ float  mrow[FQ], lrow[FQ], srow[FQ];

    const int tid  = threadIdx.x;
    const int wave = tid >> 5, lane = tid & 31;
    const int half = lane >> 4, l16 = lane & 15;
    const int wr = wave >> 1, wc = wave & 1;     // wave tile: rows wr*16, cols wc*32
    const int b = blockIdx.y >> 4;               // NH = 16
    const int h = blockIdx.y & 15;
    const int q0 = blockIdx.x * FQ;

#pragma unroll
    for (int i = 0; i < 4; ++i) {
        const int c = tid + i * 256;
        const int r = c >> 4;
        const int dc = (c & 15) << 2;
        bf16x4 v = *(const bf16x4*)(Q + (size_t)(b * Lq + q0 + r) * H_DIM + h * HEAD_D + dc);
        *(bf16x4*)&Qs[r][dc] = v;
    }
    if (tid < FQ) { mrow[tid] = -1e30f; lrow[tid] = 0.f; }

    floatx8 oacc[2] = {};
    for (int kb = 0; kb < Lk; kb += FK) {
#pragma unroll
        for (int i = 0; i < 4; ++i) {
            const int c = tid + i * 256;
            const int r = c >> 4;
            const int dc = (c & 15) << 2;
            bf16x4 kv = *(const bf16x4*)(Kb + (size_t)(b * Lk + kb + r) * H_DIM + h * HEAD_D + dc);
            *(bf16x4*)&Ks[r][dc] = kv;
            bf16x4 vv = *(const bf16x4*)(Vb + (size_t)(b * Lk + kb + r) * H_DIM + h * HEAD_D + dc);
            Vt[dc + 0][r] = vv[0]; Vt[dc + 1][r] = vv[1];
            Vt[dc + 2][r] = vv[2]; Vt[dc + 3][r] = vv[3];
        }
        __syncthreads();

        // S = Q @ K^T : 16(q) x 32(key) per wave, reduction over HD=64.
        floatx8 sacc[2] = {};
#pragma unroll
        for (int ks = 0; ks < HEAD_D; ks += 32) {
            bf16x16 aq = frag_a(&Qs[wr * 16 + l16][ks], half);
#pragma unroll
            for (int ni = 0; ni < 2; ++ni) {
                bf16x16 bk = frag_b(&Ks[wc * 32 + ni * 16 + l16][ks], half);
                sacc[ni] = wmma_bf16(aq, bk, sacc[ni]);
            }
        }
#pragma unroll
        for (int ni = 0; ni < 2; ++ni)
#pragma unroll
            for (int j = 0; j < 8; ++j)
                Ss[wr * 16 + half * 8 + j][wc * 32 + ni * 16 + l16] = sacc[ni][j] * scale;
        __syncthreads();

        if (tid < FQ) {   // streaming softmax, one row per thread
            const int r = tid;
            const float mold = mrow[r];
            float mx = mold;
            for (int j = 0; j < FK; ++j) mx = fmaxf(mx, Ss[r][j]);
            float s = 0.f;
            for (int j = 0; j < FK; ++j) {
                const float p = expf(Ss[r][j] - mx);
                s += p;
                Ps[r][j] = (__bf16)p;
            }
            const float corr = expf(mold - mx);
            mrow[r] = mx;
            lrow[r] = lrow[r] * corr + s;
            srow[r] = corr;
        }
        __syncthreads();

        // O = O*corr + P~ @ V : 16(q) x 32(hd) per wave over 64 keys.
#pragma unroll
        for (int ni = 0; ni < 2; ++ni)
#pragma unroll
            for (int j = 0; j < 8; ++j)
                oacc[ni][j] *= srow[wr * 16 + half * 8 + j];
#pragma unroll
        for (int ks = 0; ks < FK; ks += 32) {
            bf16x16 ap = frag_a(&Ps[wr * 16 + l16][ks], half);
#pragma unroll
            for (int ni = 0; ni < 2; ++ni) {
                bf16x16 bv = frag_b(&Vt[wc * 32 + ni * 16 + l16][ks], half);
                oacc[ni] = wmma_bf16(ap, bv, oacc[ni]);
            }
        }
        __syncthreads();
    }
#pragma unroll
    for (int ni = 0; ni < 2; ++ni) {
        const int col = wc * 32 + ni * 16 + l16;
#pragma unroll
        for (int j = 0; j < 8; ++j) {
            const int r = wr * 16 + half * 8 + j;
            const float inv = 1.f / lrow[r];
            O[(size_t)(b * Lq + q0 + r) * H_DIM + h * HEAD_D + col] = (__bf16)(oacc[ni][j] * inv);
        }
    }
}

// ------------------------------- Host driver --------------------------------
extern "C" void kernel_launch(void* const* d_in, const int* in_sizes, int n_in,
                              void* d_out, int out_size, void* d_ws, size_t ws_size,
                              hipStream_t stream) {
    (void)in_sizes; (void)n_in; (void)out_size; (void)ws_size;
    const float* x     = (const float*)d_in[0];
    const float* img   = (const float*)d_in[1];
    const float* W_sq  = (const float*)d_in[2];  const float* b_sq = (const float*)d_in[3];
    const float* W_sk  = (const float*)d_in[4];  const float* b_sk = (const float*)d_in[5];
    const float* W_sv  = (const float*)d_in[6];  const float* b_sv = (const float*)d_in[7];
    const float* W_so  = (const float*)d_in[8];  const float* b_so = (const float*)d_in[9];
    const float* W_cq  = (const float*)d_in[10]; const float* b_cq = (const float*)d_in[11];
    const float* W_ck  = (const float*)d_in[12]; const float* b_ck = (const float*)d_in[13];
    const float* W_cv  = (const float*)d_in[14]; const float* b_cv = (const float*)d_in[15];
    const float* W_co  = (const float*)d_in[16]; const float* b_co = (const float*)d_in[17];
    const float* W_ff1 = (const float*)d_in[18]; const float* b_ff1 = (const float*)d_in[19];
    const float* W_ff2 = (const float*)d_in[20]; const float* b_ff2 = (const float*)d_in[21];
    const float* g_ln1 = (const float*)d_in[22]; const float* b_ln1 = (const float*)d_in[23];
    const float* g_ln2 = (const float*)d_in[24]; const float* b_ln2 = (const float*)d_in[25];
    const float* g_ln3 = (const float*)d_in[26]; const float* b_ln3 = (const float*)d_in[27];

    char* base = (char*)d_ws;
    size_t off = 0;
    auto alloc = [&](size_t bytes) -> void* {
        void* p = (void*)(base + off);
        off += (bytes + 255) & ~(size_t)255;
        return p;
    };
    float*  resid = (float*) alloc((size_t)M_X * H_DIM * 4);
    __bf16* nx    = (__bf16*)alloc((size_t)M_X * H_DIM * 2);
    __bf16* qb    = (__bf16*)alloc((size_t)M_X * H_DIM * 2);
    __bf16* kb    = (__bf16*)alloc((size_t)M_X * H_DIM * 2);
    __bf16* vb    = (__bf16*)alloc((size_t)M_X * H_DIM * 2);
    __bf16* ob    = (__bf16*)alloc((size_t)M_X * H_DIM * 2);
    __bf16* imgb  = (__bf16*)alloc((size_t)M_IMG * H_DIM * 2);
    __bf16* hid   = (__bf16*)alloc((size_t)M_X * DFF_DIM * 2);
    // bf16 transposed weights Wt[N][K]
    __bf16* wt_sq = (__bf16*)alloc((size_t)H_DIM * H_DIM * 2);
    __bf16* wt_sk = (__bf16*)alloc((size_t)H_DIM * H_DIM * 2);
    __bf16* wt_sv = (__bf16*)alloc((size_t)H_DIM * H_DIM * 2);
    __bf16* wt_so = (__bf16*)alloc((size_t)H_DIM * H_DIM * 2);
    __bf16* wt_cq = (__bf16*)alloc((size_t)H_DIM * H_DIM * 2);
    __bf16* wt_ck = (__bf16*)alloc((size_t)H_DIM * H_DIM * 2);
    __bf16* wt_cv = (__bf16*)alloc((size_t)H_DIM * H_DIM * 2);
    __bf16* wt_co = (__bf16*)alloc((size_t)H_DIM * H_DIM * 2);
    __bf16* wt_f1 = (__bf16*)alloc((size_t)H_DIM * DFF_DIM * 2);
    __bf16* wt_f2 = (__bf16*)alloc((size_t)H_DIM * DFF_DIM * 2);
    float*  outF  = (float*)d_out;

    const dim3 blk(256);
    const dim3 gT(H_DIM / 32, H_DIM / 32);      // 1024x1024 transposes
    const dim3 gT1(DFF_DIM / 32, H_DIM / 32);   // W_ff1: K=1024, N=4096
    const dim3 gT2(H_DIM / 32, DFF_DIM / 32);   // W_ff2: K=4096, N=1024
    const dim3 gH(H_DIM / BN, M_X / BM);        // (8, 32)
    const dim3 gI(H_DIM / BN, M_IMG / BM);      // (8, 4)
    const dim3 gF1(DFF_DIM / BN, M_X / BM);     // (32, 32)
    const dim3 gA(SEQ / FQ, BATCH * NHEAD);     // (32, 32)

    // One-time weight transforms (bf16 + transpose) — async-copy friendly.
    transpose_cvt_kernel<<<gT,  blk, 0, stream>>>(W_sq,  wt_sq, H_DIM, H_DIM);
    transpose_cvt_kernel<<<gT,  blk, 0, stream>>>(W_sk,  wt_sk, H_DIM, H_DIM);
    transpose_cvt_kernel<<<gT,  blk, 0, stream>>>(W_sv,  wt_sv, H_DIM, H_DIM);
    transpose_cvt_kernel<<<gT,  blk, 0, stream>>>(W_so,  wt_so, H_DIM, H_DIM);
    transpose_cvt_kernel<<<gT,  blk, 0, stream>>>(W_cq,  wt_cq, H_DIM, H_DIM);
    transpose_cvt_kernel<<<gT,  blk, 0, stream>>>(W_ck,  wt_ck, H_DIM, H_DIM);
    transpose_cvt_kernel<<<gT,  blk, 0, stream>>>(W_cv,  wt_cv, H_DIM, H_DIM);
    transpose_cvt_kernel<<<gT,  blk, 0, stream>>>(W_co,  wt_co, H_DIM, H_DIM);
    transpose_cvt_kernel<<<gT1, blk, 0, stream>>>(W_ff1, wt_f1, H_DIM, DFF_DIM);
    transpose_cvt_kernel<<<gT2, blk, 0, stream>>>(W_ff2, wt_f2, DFF_DIM, H_DIM);

    // --- self attention (pre-norm) ---
    layernorm_kernel<true><<<M_X, blk, 0, stream>>>(x, resid, nx, g_ln1, b_ln1);
    scale_bf16_kernel<<<(M_IMG * H_DIM) / 256, blk, 0, stream>>>(img, imgb, M_IMG * H_DIM);
    gemm_bf16_kernel<false,false,true,false><<<gH, blk, 0, stream>>>(nx, wt_sq, b_sq, nullptr, qb, nullptr, M_X, H_DIM, H_DIM, 1.f);
    gemm_bf16_kernel<false,false,true,false><<<gH, blk, 0, stream>>>(nx, wt_sk, b_sk, nullptr, kb, nullptr, M_X, H_DIM, H_DIM, 1.f);
    gemm_bf16_kernel<false,false,true,false><<<gH, blk, 0, stream>>>(nx, wt_sv, b_sv, nullptr, vb, nullptr, M_X, H_DIM, H_DIM, 1.f);
    flash_attn_kernel<<<gA, blk, 0, stream>>>(qb, kb, vb, ob, SEQ, SEQ, 0.125f);
    gemm_bf16_kernel<false,true,false,true><<<gH, blk, 0, stream>>>(ob, wt_so, b_so, resid, nullptr, resid, M_X, H_DIM, H_DIM, 1.f);

    // --- cross attention (pre-norm) ---
    layernorm_kernel<false><<<M_X, blk, 0, stream>>>(resid, nullptr, nx, g_ln2, b_ln2);
    gemm_bf16_kernel<false,false,true,false><<<gH, blk, 0, stream>>>(nx, wt_cq, b_cq, nullptr, qb, nullptr, M_X, H_DIM, H_DIM, 1.f);
    gemm_bf16_kernel<false,false,true,false><<<gI, blk, 0, stream>>>(imgb, wt_ck, b_ck, nullptr, kb, nullptr, M_IMG, H_DIM, H_DIM, 1.f);
    gemm_bf16_kernel<false,false,true,false><<<gI, blk, 0, stream>>>(imgb, wt_cv, b_cv, nullptr, vb, nullptr, M_IMG, H_DIM, H_DIM, 1.f);
    flash_attn_kernel<<<gA, blk, 0, stream>>>(qb, kb, vb, ob, SEQ, SEQ_I, 0.125f);
    gemm_bf16_kernel<false,true,false,true><<<gH, blk, 0, stream>>>(ob, wt_co, b_co, resid, nullptr, resid, M_X, H_DIM, H_DIM, 1.f);

    // --- feed forward (pre-norm), final *10 folded into ff2 epilogue ---
    layernorm_kernel<false><<<M_X, blk, 0, stream>>>(resid, nullptr, nx, g_ln3, b_ln3);
    gemm_bf16_kernel<true,false,true,false><<<gF1, blk, 0, stream>>>(nx, wt_f1, b_ff1, nullptr, hid, nullptr, M_X, DFF_DIM, H_DIM, 1.f);
    gemm_bf16_kernel<false,true,false,true><<<gH, blk, 0, stream>>>(hid, wt_f2, b_ff2, resid, nullptr, outF, M_X, H_DIM, DFF_DIM, 10.f);
}
